// CNNEmbedding_25855703122226
// MI455X (gfx1250) — matrix operand order repaired and checked by reading, not compiled, not run on hardware
//
#include <hip/hip_runtime.h>
#include <hip/hip_bf16.h>

typedef unsigned int   u32;
typedef unsigned short u16;

typedef __attribute__((ext_vector_type(16))) __bf16 v16bf;
typedef __attribute__((ext_vector_type(8)))  float  v8f;

union Frag32B { uint4 q[2]; v16bf v; };

// ---------------- tile tables ----------------
// 36 channel tiles of 16 (padded): w=1..6 -> 2,4,5,7,8,10 tiles
__constant__ int TILE_CONV[36] = {0,0, 1,1,1,1, 2,2,2,2,2,
                                  3,3,3,3,3,3,3, 4,4,4,4,4,4,4,4,
                                  5,5,5,5,5,5,5,5,5,5};
__constant__ int TILE_TJ[36]   = {0,1, 0,1,2,3, 0,1,2,3,4,
                                  0,1,2,3,4,5,6, 0,1,2,3,4,5,6,7,
                                  0,1,2,3,4,5,6,7,8,9};
// cumulative fragment index (each tile has 2*w fragments of 512 bf16 = 1KB)
__constant__ int TILE_FRAG[36] = {0,2, 4,8,12,16, 20,26,32,38,44,
                                  50,58,66,74,82,90,98,
                                  106,116,126,136,146,156,166,176,
                                  186,198,210,222,234,246,258,270,282,294};
__constant__ int CONV_C[6]    = {25,50,75,100,125,150};
__constant__ int CONV_BASE[6] = {0,25,75,150,250,375};

// balanced static tile->wave partition (cost = T_w * 2w per tile):
// wave costs {1516,1516,1468,1560} vs ideal 1515
__constant__ int WAVE_TILES[36] = {
  26,27,28, 18,19, 11, 6, 2, 0,          // wave 0 (9 tiles)
  29,30,31, 20,21, 12, 7, 3, 1,          // wave 1 (9 tiles)
  32,33, 22,23, 13,14,15, 8,             // wave 2 (8 tiles)
  34,35, 24,25, 16,17, 9,10, 4,5         // wave 3 (10 tiles)
};
__constant__ int WAVE_START[5] = {0, 9, 18, 26, 36};

struct Params6 { const float* p[6]; };

__device__ __forceinline__ u32 f2bf(float x) {          // RNE fp32 -> bf16 bits
  u32 u = __float_as_uint(x);
  return (u + 0x7fffu + ((u >> 16) & 1u)) >> 16;
}

// ---------------- weight prepack: fp32 [C,64,w] -> bf16 A-fragments ----------------
// A 16x32 bf16 layout: lane l holds M = l%16; elem e -> K = ks*32 + (e<8?e:e+8) + (l>=16?8:0)
__global__ void prepack_weights(Params6 Ws, u16* __restrict__ frag) {
  int f    = blockIdx.x;       // fragment id 0..305
  int lane = threadIdx.x;      // 0..31
  int ti = 0;
  while (ti + 1 < 36 && TILE_FRAG[ti + 1] <= f) ++ti;
  int ks = f - TILE_FRAG[ti];
  int cv = TILE_CONV[ti];
  int KW = cv + 1;
  int C  = CONV_C[cv];
  const float* Wp = Ws.p[cv];
  int c  = TILE_TJ[ti] * 16 + (lane & 15);
  int kb = (lane >> 4) * 8;
  u32* dst = (u32*)(frag + (size_t)f * 512 + lane * 16);
  #pragma unroll
  for (int e = 0; e < 16; e += 2) {
    int K0 = ks * 32 + ((e     < 8) ? e     : e + 8) + kb;
    int K1 = ks * 32 + ((e + 1 < 8) ? e + 1 : e + 9) + kb;
    float a = (c < C) ? Wp[(c * 64 + (K0 & 63)) * KW + (K0 >> 6)] : 0.f;
    float b = (c < C) ? Wp[(c * 64 + (K1 & 63)) * KW + (K1 >> 6)] : 0.f;
    dst[e >> 1] = f2bf(a) | (f2bf(b) << 16);
  }
}

// ---------------- per-tile WMMA conv + max-pool ----------------
// sX layout: [tp 0..25][word 0..15][d 0..71(pad)], bf16. tp = time + 5, zeros outside [5,21)
template<int KW>
__device__ __forceinline__ void doTile(const u16* sX, const u16* frag, const float* bias,
                                       float* out, int chanBase, int Cc, int tj,
                                       int lane, int group) {
  constexpr int NK = 2 * KW;
  constexpr int T  = 15 + KW;
  v16bf A[NK];
  #pragma unroll
  for (int ks = 0; ks < NK; ++ks) {
    const uint4* p = (const uint4*)(frag + ks * 512 + lane * 16);
    Frag32B u; u.q[0] = p[0]; u.q[1] = p[1];
    A[ks] = u.v;
  }
  const int word  = lane & 15;
  const int halfK = lane >> 4;
  const u16* laneBase = sX + word * 72 + halfK * 8;    // per-lane invariant base
  v8f mx;
  #pragma unroll
  for (int i = 0; i < 8; ++i) mx[i] = -3.402823466e38f;

  for (int t = 0; t < T; ++t) {
    v8f acc0 = {}, acc1 = {};                          // two independent WMMA chains
    #pragma unroll
    for (int j = 0; j < KW; ++j) {                     // conv tap j: K rows 64j..64j+63
      int tp = t + j - (KW - 1) + 5;                   // padded time row, in [0,26)
      const uint4* bp = (const uint4*)(laneBase + tp * 1152);
      Frag32B b0; b0.q[0] = bp[0]; b0.q[1] = bp[2];    // d = h*8 .. +7, 16+h*8 .. +7
      Frag32B b1; b1.q[0] = bp[4]; b1.q[1] = bp[6];    // d = 32+h*8 .. +7, 48+h*8 .. +7
      acc0 = __builtin_amdgcn_wmma_f32_16x16x32_bf16(
               false, A[2 * j],     false, b0.v, (short)0, acc0, false, false);
      acc1 = __builtin_amdgcn_wmma_f32_16x16x32_bf16(
               false, A[2 * j + 1], false, b1.v, (short)0, acc1, false, false);
    }
    #pragma unroll
    for (int i = 0; i < 8; ++i) mx[i] = fmaxf(mx[i], fmaxf(acc0[i], acc1[i]));
  }
  // epilogue: D layout -> lane, vgpr v: channel = v + 8*halfK, word = lane%16
  int wordG = group * 16 + word;
  int pos = wordG & 255, batch = wordG >> 8;
  float* orow = out + ((size_t)pos * 64 + batch) * 525;
  #pragma unroll
  for (int v = 0; v < 8; ++v) {
    int cc = tj * 16 + v + 8 * halfK;                  // channel within this conv
    if (cc < Cc) orow[chanBase + cc] = tanhf(mx[v] + bias[cc]);
  }
}

// ---------------- main fused kernel: 16 words per block ----------------
__global__ __launch_bounds__(128) void char_cnn_kernel(
    const int* __restrict__ cid, const float* __restrict__ emb,
    const u16* __restrict__ frag, Params6 Bs, float* __restrict__ out) {
  __shared__ __align__(16) u16 sX[26 * 16 * 72];      // 59,904 bytes
  const int tid = threadIdx.x, g = blockIdx.x;

  // zero the time-padding rows (tp 0..4 and 21..25)
  u32* z = (u32*)sX;
  for (int i = tid; i < 2880; i += 128) { z[i] = 0; z[12096 + i] = 0; }

  // gather + convert: 16 words x 16 chars, 64 dims each
  for (int p = tid; p < 256; p += 128) {
    int word = p & 15, ch = p >> 4;
    int id = cid[(g * 16 + word) * 16 + ch];
    const float4* e = (const float4*)(emb + (size_t)id * 64);
    u16* dst = sX + ((ch + 5) * 16 + word) * 72;
    #pragma unroll
    for (int q = 0; q < 16; ++q) {
      float4 fv = e[q];
      uint2 pk;
      pk.x = f2bf(fv.x) | (f2bf(fv.y) << 16);
      pk.y = f2bf(fv.z) | (f2bf(fv.w) << 16);
      *(uint2*)(dst + q * 4) = pk;
    }
  }
  __syncthreads();

  const int wave = tid >> 5, lane = tid & 31;
  const int i0 = WAVE_START[wave], i1 = WAVE_START[wave + 1];
  for (int i = i0; i < i1; ++i) {                      // uniform per wave -> EXEC all ones
    int ti = WAVE_TILES[i];
    int cv = TILE_CONV[ti];
    const u16* fb = frag + (size_t)TILE_FRAG[ti] * 512;
    const float* bias = Bs.p[cv];
    int tj = TILE_TJ[ti], cb = CONV_BASE[cv], Cc = CONV_C[cv];
    switch (cv) {
      case 0: doTile<1>(sX, fb, bias, out, cb, Cc, tj, lane, g); break;
      case 1: doTile<2>(sX, fb, bias, out, cb, Cc, tj, lane, g); break;
      case 2: doTile<3>(sX, fb, bias, out, cb, Cc, tj, lane, g); break;
      case 3: doTile<4>(sX, fb, bias, out, cb, Cc, tj, lane, g); break;
      case 4: doTile<5>(sX, fb, bias, out, cb, Cc, tj, lane, g); break;
      case 5: doTile<6>(sX, fb, bias, out, cb, Cc, tj, lane, g); break;
    }
  }
}

extern "C" void kernel_launch(void* const* d_in, const int* in_sizes, int n_in,
                              void* d_out, int out_size, void* d_ws, size_t ws_size,
                              hipStream_t stream) {
  (void)in_sizes; (void)n_in; (void)out_size; (void)ws_size;
  const int*   cid = (const int*)d_in[0];
  const float* emb = (const float*)d_in[3];
  Params6 Ws, Bs;
  for (int i = 0; i < 6; ++i) {
    Ws.p[i] = (const float*)d_in[4 + 2 * i];
    Bs.p[i] = (const float*)d_in[5 + 2 * i];
  }
  u16* frag = (u16*)d_ws;                              // needs 306 KB
  prepack_weights<<<306, 32, 0, stream>>>(Ws, frag);   // fp32 weights -> bf16 A-fragments
  char_cnn_kernel<<<1024, 128, 0, stream>>>(cid, emb, frag, Bs, (float*)d_out);
}
